// LSTM_13159779795397
// MI455X (gfx1250) — matrix operand order repaired and checked by reading, not compiled, not run on hardware
//
#include <hip/hip_runtime.h>

// Problem constants (from reference)
#define B_     64
#define T_     512
#define DIN_   256
#define H_     1024
#define G4H_   4096
#define DOUT_  256
#define CHUNK_ 64
#define NCHUNK_ (T_ / CHUNK_)
#define STEP_BLOCKS_ 32           // persistent kernel grid
#define LDS_W_BYTES (2 * 4 * 16 * H_ * 2)   // 256 KB: 2 j-slots x 4 gates x 16 rows x H bf16

typedef unsigned short u16b; // raw bf16 storage
typedef __attribute__((ext_vector_type(16))) __bf16        v16bf;
typedef __attribute__((ext_vector_type(8)))  float         v8f;
typedef __attribute__((ext_vector_type(4)))  unsigned int  v4u;

union FragCaster {
    v16bf bf;
    v4u   q[2];
};

__device__ __forceinline__ u16b f32_to_bf16(float f) {
    unsigned int u = __float_as_uint(f);
    unsigned int r = 0x7FFFu + ((u >> 16) & 1u);
    return (u16b)((u + r) >> 16);
}

// A fragment (16x32 bf16, MxK): lane<16 -> row=lane, K = {0..7, 16..23};
// lane>=16 -> row=lane-16, K = {8..15, 24..31}. Two 16B loads at p, p+16 elems.
__device__ __forceinline__ v16bf load_frag_a(const u16b* p) {
    FragCaster f;
    f.q[0] = *(const v4u*)(p);
    f.q[1] = *(const v4u*)(p + 16);
    return f.bf;
}

// B fragment (32x16 bf16, KxN): lane<16 -> col=lane, K = {0..7, 16..23};
// lane>=16 -> col=lane-16, K = {8..15, 24..31}. Two 16B loads at p, p+16 elems.
__device__ __forceinline__ v16bf load_frag_b(const u16b* p) {
    FragCaster f;
    f.q[0] = *(const v4u*)(p);
    f.q[1] = *(const v4u*)(p + 16);
    return f.bf;
}

__device__ __forceinline__ v8f wmma_bf16(v16bf a, v16bf b, v8f c) {
    return __builtin_amdgcn_wmma_f32_16x16x32_bf16(
        false, a, false, b, (short)0, c, false, false);
}

__device__ __forceinline__ float sigmoidf_(float x) {
    return 1.0f / (1.0f + __expf(-x));
}

// ---------------------------------------------------------------- elementwise
__global__ void k_cvt_bf16(const float* __restrict__ src, u16b* __restrict__ dst, int n) {
    for (int i = blockIdx.x * blockDim.x + threadIdx.x; i < n; i += gridDim.x * blockDim.x)
        dst[i] = f32_to_bf16(src[i]);
}

__global__ void k_add_bias(const float* __restrict__ a, const float* __restrict__ b,
                           float* __restrict__ dst, int n) {
    for (int i = blockIdx.x * blockDim.x + threadIdx.x; i < n; i += gridDim.x * blockDim.x)
        dst[i] = a[i] + b[i];
}

__global__ void k_zero_bf16(u16b* __restrict__ p, int n) {
    for (int i = blockIdx.x * blockDim.x + threadIdx.x; i < n; i += gridDim.x * blockDim.x)
        p[i] = 0;
}

__global__ void k_zero_f32(float* __restrict__ p, int n) {
    for (int i = blockIdx.x * blockDim.x + threadIdx.x; i < n; i += gridDim.x * blockDim.x)
        p[i] = 0.0f;
}

__global__ void k_zero_u32(unsigned* __restrict__ p, int n) {
    for (int i = blockIdx.x * blockDim.x + threadIdx.x; i < n; i += gridDim.x * blockDim.x)
        p[i] = 0u;
}

// Pack x[b, ct0+tl, k] (f32, [B,T,DIN]) -> A0[(tl*B+b)*DIN + k] (bf16)
__global__ void k_pack_x(const float* __restrict__ x, u16b* __restrict__ dst, int ct0) {
    const int n = CHUNK_ * B_ * DIN_;
    for (int i = blockIdx.x * blockDim.x + threadIdx.x; i < n; i += gridDim.x * blockDim.x) {
        int k  = i % DIN_;
        int b  = (i / DIN_) % B_;
        int tl = i / (DIN_ * B_);
        dst[i] = f32_to_bf16(x[((size_t)b * T_ + ct0 + tl) * DIN_ + k]);
    }
}

// ---------------------------------------------------------------- xg GEMM
// out[m, n] = sum_k A[m,k] * W[n,k] + bias[n]; M = CHUNK_*B_ = 4096, N = 4096.
// One wave computes a 16x64 strip (4 accumulators), reusing the A fragment 4x.
__global__ void k_gemm_xg(const u16b* __restrict__ A, const u16b* __restrict__ W,
                          const float* __restrict__ bias, float* __restrict__ out,
                          int M, int K) {
    const int lane = threadIdx.x & 31;
    const int wid  = blockIdx.x * (blockDim.x >> 5) + (threadIdx.x >> 5);
    const int mtiles = M >> 4;
    const int mt = wid % mtiles;
    const int ng = wid / mtiles;
    if (ng >= (G4H_ / 64)) return;
    const int m0 = mt << 4, n0 = ng << 6;
    const int half = lane >> 4, r = lane & 15;

    v8f acc[4] = {};
    const u16b* arow = A + (size_t)(m0 + r) * K;
    for (int k0 = 0; k0 < K; k0 += 32) {
        v16bf af = load_frag_a(arow + k0 + half * 8);
        __builtin_prefetch(arow + k0 + 64, 0, 1);
#pragma unroll
        for (int nb = 0; nb < 4; ++nb) {
            const u16b* wrow = W + (size_t)(n0 + nb * 16 + r) * K + k0 + half * 8;
            v16bf bf = load_frag_b(wrow);
            acc[nb] = wmma_bf16(af, bf, acc[nb]);
        }
    }
#pragma unroll
    for (int nb = 0; nb < 4; ++nb) {
#pragma unroll
        for (int i = 0; i < 8; ++i) {
            int row = m0 + i + 8 * half;
            int col = n0 + nb * 16 + r;
            out[(size_t)row * G4H_ + col] = acc[nb][i] + bias[col];
        }
    }
}

// ---------------------------------------------------------------- persistent LSTM chunk
// Runs CHUNK_ timesteps in one launch. 32 blocks x 256 thr = 256 waves =
// (B_/16)*(H_/16) tiles; each wave owns a fixed 16x16 (batch x hidden) block and
// computes all four gate tiles per step. Whh slice for the block lives in LDS
// (256 KB of the 320 KB/WGP). Cell state stays in registers for the whole chunk.
// Steps are separated by an agent-scope atomic grid barrier (monotonic counter).
__global__ void __launch_bounds__(256) k_lstm_chunk(
        const float* __restrict__ xg,       // [CHUNK_, B_, 4H] f32
        const u16b*  __restrict__ Whh,      // [4H, H] bf16
        float*       __restrict__ c,        // [B_, H] f32 (in/out)
        u16b*        __restrict__ ha,       // h ping-pong buffer A (read at even t)
        u16b*        __restrict__ hbuf,     // h ping-pong buffer B
        u16b*        __restrict__ seq_out,  // [CHUNK_, B_, H] bf16 or nullptr
        unsigned*    __restrict__ bar_cnt,  // zeroed before launch
        int ct0)
{
    extern __shared__ __align__(16) u16b ldsW[];   // [2 slots][4 gates][16 rows][H_]
    const int lane = threadIdx.x & 31;
    const int w    = threadIdx.x >> 5;
    const int wid  = blockIdx.x * 8 + w;
    const int mt = wid & 3;        // B_/16 = 4
    const int jt = wid >> 2;       // H_/16 = 64  (block covers jt = blockIdx*2 + {0,1})
    const int m0 = mt << 4, j0 = jt << 4;
    const int half = lane >> 4, r = lane & 15;
    const int slot = w >> 2;       // which j-slot of this block this wave uses

    // Stage the block's Whh rows into LDS (fragment-ready, row-major).
    const int jt0 = blockIdx.x * 2;
    for (int idx = threadIdx.x; idx < (2 * 4 * 16 * H_) / 8; idx += blockDim.x) {
        const int elem    = idx * 8;
        const int row_lin = elem / H_;
        const int koff    = elem - row_lin * H_;
        const int s  = row_lin >> 6;
        const int g  = (row_lin >> 4) & 3;
        const int rr = row_lin & 15;
        *(v4u*)(ldsW + (size_t)row_lin * H_ + koff) =
            *(const v4u*)(Whh + (size_t)(g * H_ + (jt0 + s) * 16 + rr) * H_ + koff);
    }
    __syncthreads();

    // Cell-state tile in registers for the whole chunk.
    float creg[8];
#pragma unroll
    for (int i = 0; i < 8; ++i)
        creg[i] = c[(size_t)(m0 + i + 8 * half) * H_ + (j0 + r)];

    const u16b* ldsB = ldsW + (size_t)slot * (4 * 16 * H_);

    for (int tl = 0; tl < CHUNK_; ++tl) {
        const int t = ct0 + tl;
        const u16b* hin  = (t & 1) ? hbuf : ha;
        u16b*       hout = (t & 1) ? ha   : hbuf;
        const float* xg_t = xg + (size_t)tl * B_ * G4H_;

        v8f acc[4] = {};
        const u16b* arow = hin + (size_t)(m0 + r) * H_;
        for (int k0 = 0; k0 < H_; k0 += 32) {
            v16bf af = load_frag_a(arow + k0 + half * 8);
#pragma unroll
            for (int g = 0; g < 4; ++g) {
                v16bf bf = load_frag_b(ldsB + (size_t)(g * 16 + r) * H_ + k0 + half * 8);
                acc[g] = wmma_bf16(af, bf, acc[g]);
            }
        }

#pragma unroll
        for (int i = 0; i < 8; ++i) {
            const int brow = m0 + i + 8 * half;
            const int col  = j0 + r;
            const float* xr = xg_t + (size_t)brow * G4H_;
            const float ipre = acc[0][i] + xr[0 * H_ + col];
            const float fpre = acc[1][i] + xr[1 * H_ + col];
            const float gpre = acc[2][i] + xr[2 * H_ + col];
            const float opre = acc[3][i] + xr[3 * H_ + col];
            const float ig = sigmoidf_(ipre);
            const float fg = sigmoidf_(fpre);
            const float gg = tanhf(gpre);
            const float og = sigmoidf_(opre);
            creg[i] = fg * creg[i] + ig * gg;
            const u16b hv = f32_to_bf16(og * tanhf(creg[i]));
            hout[(size_t)brow * H_ + col] = hv;
            if (seq_out)
                seq_out[(size_t)tl * B_ * H_ + (size_t)brow * H_ + col] = hv;
        }

        if (tl != CHUNK_ - 1) {
            __threadfence();           // make this thread's h stores agent-visible
            __syncthreads();           // whole block has fenced
            if (threadIdx.x == 0) {
                __hip_atomic_fetch_add(bar_cnt, 1u, __ATOMIC_RELEASE,
                                       __HIP_MEMORY_SCOPE_AGENT);
                const unsigned target = (unsigned)(tl + 1) * (unsigned)gridDim.x;
                while (__hip_atomic_load(bar_cnt, __ATOMIC_ACQUIRE,
                                         __HIP_MEMORY_SCOPE_AGENT) < target)
                    __builtin_amdgcn_s_sleep(1);
            }
            __syncthreads();
        }
    }

#pragma unroll
    for (int i = 0; i < 8; ++i)
        c[(size_t)(m0 + i + 8 * half) * H_ + (j0 + r)] = creg[i];
}

// ---------------------------------------------------------------- final projection
// out[r, n] = Hcat[r, :] @ Wout[n, :] + bout[n]; M=128 (h0|h1), N=256, K=1024.
__global__ void k_final_gemm(const u16b* __restrict__ h0, const u16b* __restrict__ h1,
                             const u16b* __restrict__ Wo, const float* __restrict__ bout,
                             float* __restrict__ out) {
    const int lane = threadIdx.x & 31;
    const int wid  = blockIdx.x * (blockDim.x >> 5) + (threadIdx.x >> 5);
    const int mt = wid & 7;        // 128/16 = 8
    const int nt = wid >> 3;       // 256/16 = 16
    if (nt >= 16) return;
    const int m0 = mt << 4, n0 = nt << 4;
    const int half = lane >> 4, r = lane & 15;

    const u16b* Abase = (m0 < B_) ? (h0 + (size_t)m0 * H_) : (h1 + (size_t)(m0 - B_) * H_);
    v8f acc = {};
    const u16b* arow = Abase + (size_t)r * H_;
    for (int k0 = 0; k0 < H_; k0 += 32) {
        v16bf af = load_frag_a(arow + k0 + half * 8);
        v16bf bf = load_frag_b(Wo + (size_t)(n0 + r) * H_ + k0 + half * 8);
        acc = wmma_bf16(af, bf, acc);
    }
#pragma unroll
    for (int i = 0; i < 8; ++i) {
        int row = m0 + i + 8 * half;
        int col = n0 + r;
        out[(size_t)row * DOUT_ + col] = acc[i] + bout[col];
    }
}

// ---------------------------------------------------------------- launcher
static inline int ceil_div(int a, int b) { return (a + b - 1) / b; }

extern "C" void kernel_launch(void* const* d_in, const int* in_sizes, int n_in,
                              void* d_out, int out_size, void* d_ws, size_t ws_size,
                              hipStream_t stream) {
    (void)in_sizes; (void)n_in; (void)out_size; (void)ws_size;

    const float* x    = (const float*)d_in[0];
    const float* Wih0 = (const float*)d_in[1];
    const float* Whh0 = (const float*)d_in[2];
    const float* bih0 = (const float*)d_in[3];
    const float* bhh0 = (const float*)d_in[4];
    const float* Wih1 = (const float*)d_in[5];
    const float* Whh1 = (const float*)d_in[6];
    const float* bih1 = (const float*)d_in[7];
    const float* bhh1 = (const float*)d_in[8];
    const float* Wout = (const float*)d_in[9];
    const float* bout = (const float*)d_in[10];
    float* out = (float*)d_out;

    char* base = (char*)d_ws;
    size_t off = 0;
    auto carve = [&](size_t bytes) -> char* {
        char* p = base + off;
        off += (bytes + 255) & ~(size_t)255;
        return p;
    };

    u16b* Wih0b = (u16b*)carve((size_t)G4H_ * DIN_ * 2);
    u16b* Whh0b = (u16b*)carve((size_t)G4H_ * H_ * 2);
    u16b* Wih1b = (u16b*)carve((size_t)G4H_ * H_ * 2);
    u16b* Whh1b = (u16b*)carve((size_t)G4H_ * H_ * 2);
    u16b* Woutb = (u16b*)carve((size_t)DOUT_ * H_ * 2);
    float* bias0 = (float*)carve((size_t)G4H_ * 4);
    float* bias1 = (float*)carve((size_t)G4H_ * 4);
    u16b* A0    = (u16b*)carve((size_t)CHUNK_ * B_ * DIN_ * 2);
    u16b* seq0  = (u16b*)carve((size_t)CHUNK_ * B_ * H_ * 2);
    float* xg   = (float*)carve((size_t)CHUNK_ * B_ * G4H_ * 4);   // shared by both layers
    u16b* h0a = (u16b*)carve((size_t)B_ * H_ * 2);
    u16b* h0b = (u16b*)carve((size_t)B_ * H_ * 2);
    u16b* h1a = (u16b*)carve((size_t)B_ * H_ * 2);
    u16b* h1b = (u16b*)carve((size_t)B_ * H_ * 2);
    float* c0 = (float*)carve((size_t)B_ * H_ * 4);
    float* c1 = (float*)carve((size_t)B_ * H_ * 4);
    unsigned* counters = (unsigned*)carve(2 * NCHUNK_ * sizeof(unsigned));

    // Weight / bias conversion (every call; deterministic)
    k_cvt_bf16<<<ceil_div(G4H_ * DIN_, 256), 256, 0, stream>>>(Wih0, Wih0b, G4H_ * DIN_);
    k_cvt_bf16<<<ceil_div(G4H_ * H_, 256), 256, 0, stream>>>(Whh0, Whh0b, G4H_ * H_);
    k_cvt_bf16<<<ceil_div(G4H_ * H_, 256), 256, 0, stream>>>(Wih1, Wih1b, G4H_ * H_);
    k_cvt_bf16<<<ceil_div(G4H_ * H_, 256), 256, 0, stream>>>(Whh1, Whh1b, G4H_ * H_);
    k_cvt_bf16<<<ceil_div(DOUT_ * H_, 256), 256, 0, stream>>>(Wout, Woutb, DOUT_ * H_);
    k_add_bias<<<ceil_div(G4H_, 256), 256, 0, stream>>>(bih0, bhh0, bias0, G4H_);
    k_add_bias<<<ceil_div(G4H_, 256), 256, 0, stream>>>(bih1, bhh1, bias1, G4H_);

    // Initial state: h(t=0)=0 in buffer 'a' (read at even t), c=0, barriers=0.
    k_zero_bf16<<<ceil_div(B_ * H_, 256), 256, 0, stream>>>(h0a, B_ * H_);
    k_zero_bf16<<<ceil_div(B_ * H_, 256), 256, 0, stream>>>(h1a, B_ * H_);
    k_zero_f32<<<ceil_div(B_ * H_, 256), 256, 0, stream>>>(c0, B_ * H_);
    k_zero_f32<<<ceil_div(B_ * H_, 256), 256, 0, stream>>>(c1, B_ * H_);
    k_zero_u32<<<1, 64, 0, stream>>>(counters, 2 * NCHUNK_);

    const int gemm_blocks = ((CHUNK_ * B_ / 16) * (G4H_ / 64)) / 8;  // 2048

    for (int ch = 0; ch < NCHUNK_; ++ch) {
        const int ct0 = ch * CHUNK_;

        // Layer 0: input GEMM for this chunk, then one persistent 64-step kernel
        k_pack_x<<<ceil_div(CHUNK_ * B_ * DIN_, 256), 256, 0, stream>>>(x, A0, ct0);
        k_gemm_xg<<<gemm_blocks, 256, 0, stream>>>(A0, Wih0b, bias0, xg, CHUNK_ * B_, DIN_);
        k_lstm_chunk<<<STEP_BLOCKS_, 256, LDS_W_BYTES, stream>>>(
            xg, Whh0b, c0, h0a, h0b, seq0, counters + (ch * 2 + 0), ct0);

        // Layer 1: input GEMM on layer-0 hidden chunk, then persistent steps
        k_gemm_xg<<<gemm_blocks, 256, 0, stream>>>(seq0, Wih1b, bias1, xg, CHUNK_ * B_, H_);
        k_lstm_chunk<<<STEP_BLOCKS_, 256, LDS_W_BYTES, stream>>>(
            xg, Whh1b, c1, h1a, h1b, (u16b*)nullptr, counters + (ch * 2 + 1), ct0);
    }

    // After t=511 (odd), final h lives in buffer 'a' for both layers.
    k_final_gemm<<<16, 256, 0, stream>>>(h0a, h1a, Woutb, bout, out);
}
